// TPGroupQuantLinear_11733850653105
// MI455X (gfx1250) — matrix operand order repaired and checked by reading, not compiled, not run on hardware
//
#include <hip/hip_runtime.h>

typedef __attribute__((ext_vector_type(16))) _Float16 v16h;
typedef __attribute__((ext_vector_type(8)))  _Float16 v8h;
typedef __attribute__((ext_vector_type(2)))  _Float16 v2h;
typedef __attribute__((ext_vector_type(8)))  float    v8f;
typedef __attribute__((ext_vector_type(4)))  float    v4f;
typedef __attribute__((ext_vector_type(4)))  int      v4i;
typedef __attribute__((ext_vector_type(4)))  unsigned v4u;

// Address-space-qualified pointers for the async-to-LDS builtin
// (signature: (AS1 v4i* src_global, AS3 v4i* dst_lds, imm offset, imm cpol)).
typedef __attribute__((address_space(1))) v4i* gas_v4i;
typedef __attribute__((address_space(3))) v4i* las_v4i;

#define K_DIM          4096
#define N_DIM          4096
#define GROUPS         64
#define GK             64
#define WORDS_PER_ROW  1024
#define BM             128
#define BN             128
#define F16_STRIDE     72      // halfs per row: 64 + 8 pad
#define F32_STRIDE     68      // floats per row: 64 + 4 pad (272B, 16B aligned)

#if defined(__AMDGCN__) && defined(__gfx1250__) && \
    __has_builtin(__builtin_amdgcn_global_load_async_to_lds_b128) && \
    __has_builtin(__builtin_amdgcn_s_wait_asynccnt)
#define HAVE_ASYNC 1
#else
#define HAVE_ASYNC 0
#endif

// int4 -> f16 dequant, bit-trick version:
//   0x6400 | q  is the f16 bit pattern of (1024 + q)  (q in [0,15], exact)
//   (h - 1024) recovers q exactly in f16, then one packed fma applies scale/bias.
__device__ __forceinline__ v16h dequant16(v4i w, v2h s2, v2h b2) {
    const v2h k1024 = {(_Float16)1024.0f, (_Float16)1024.0f};
    v16h r;
#pragma unroll
    for (int i = 0; i < 4; ++i) {
        unsigned t  = (unsigned)w[i];
        unsigned p0 = ((t & 0x0000000Fu) | ((t << 12) & 0x000F0000u)) | 0x64006400u;
        unsigned p1 = (((t >> 8) & 0x0000000Fu) | ((t << 4) & 0x000F0000u)) | 0x64006400u;
        v2h q0 = __builtin_bit_cast(v2h, p0) - k1024;   // {q0, q1} exact
        v2h q1 = __builtin_bit_cast(v2h, p1) - k1024;   // {q2, q3} exact
        v2h r0 = q0 * s2 + b2;                          // v_pk_fma_f16
        v2h r1 = q1 * s2 + b2;
        r[4 * i + 0] = r0.x; r[4 * i + 1] = r0.y;
        r[4 * i + 2] = r1.x; r[4 * i + 3] = r1.y;
    }
    return r;
}

__global__ __launch_bounds__(256)
void gq_linear_wmma(const float* __restrict__ x,
                    const int*   __restrict__ wp,
                    const float* __restrict__ ws,
                    const float* __restrict__ wb,
                    const float* __restrict__ bias,
                    float* __restrict__ out)
{
    __shared__ _Float16 xh[BM * F16_STRIDE];          // 18 KB f16 WMMA tile
#if HAVE_ASYNC
    __shared__ float xs32[2][BM * F32_STRIDE];        // 2 x 34.8 KB f32 staging
#endif

    const int tid   = threadIdx.x;
    const int wave  = tid >> 5;
    const int lane  = tid & 31;
    const int lmod  = lane & 15;
    const int lhalf = lane >> 4;

    const int wm = wave & 1;      // M sub-block (2 x 64 rows)
    const int wn = wave >> 1;     // N sub-block (4 x 32 cols)

    const int mblk = blockIdx.y * BM;
    const int nblk = blockIdx.x * BN;

    v8f acc[4][2];
#pragma unroll
    for (int mt = 0; mt < 4; ++mt)
#pragma unroll
        for (int nt = 0; nt < 2; ++nt)
            acc[mt][nt] = (v8f){0.f, 0.f, 0.f, 0.f, 0.f, 0.f, 0.f, 0.f};

#if HAVE_ASYNC
    // Issue one 128x64 f32 tile (32 KB) as 8 async B128 copies per thread.
    auto issue_tile = [&](int g, int buf) {
        const int kb = g * GK;
#pragma unroll
        for (int i = 0; i < 8; ++i) {
            int seg = tid + i * 256;
            int row = seg >> 4;
            int c4  = (seg & 15) << 2;
            const float* gp = &x[(size_t)(mblk + row) * K_DIM + kb + c4];
            float*       lp = &xs32[buf][row * F32_STRIDE + c4];
            __builtin_amdgcn_global_load_async_to_lds_b128(
                (gas_v4i)gp, (las_v4i)lp, 0, 0);
        }
    };
    issue_tile(0, 0);
#endif

    for (int g = 0; g < GROUPS; ++g) {
#if HAVE_ASYNC
        const int cur = g & 1;
        if (g + 1 < GROUPS) {
            issue_tile(g + 1, cur ^ 1);                 // prefetch next tile
            __builtin_amdgcn_s_wait_asynccnt(8);        // this tile's 8 copies done
        } else {
            __builtin_amdgcn_s_wait_asynccnt(0);
        }
        __syncthreads();
        // LDS f32 -> LDS f16 conversion pass: 8-float chunks, b128 stores.
        // 128 rows x 8 chunks = 1024 chunks / 256 threads = 4 per thread.
#pragma unroll
        for (int i = 0; i < 4; ++i) {
            int ch  = tid + i * 256;
            int row = ch >> 3;
            int c8  = (ch & 7) << 3;
            const float* sp = &xs32[cur][row * F32_STRIDE + c8];
            v4f v0 = *reinterpret_cast<const v4f*>(sp);
            v4f v1 = *reinterpret_cast<const v4f*>(sp + 4);
            v4u u;
            u.x = __builtin_bit_cast(unsigned, __builtin_amdgcn_cvt_pkrtz(v0.x, v0.y));
            u.y = __builtin_bit_cast(unsigned, __builtin_amdgcn_cvt_pkrtz(v0.z, v0.w));
            u.z = __builtin_bit_cast(unsigned, __builtin_amdgcn_cvt_pkrtz(v1.x, v1.y));
            u.w = __builtin_bit_cast(unsigned, __builtin_amdgcn_cvt_pkrtz(v1.z, v1.w));
            *reinterpret_cast<v4u*>(&xh[row * F16_STRIDE + c8]) = u;
        }
#else
        const int kbase = g * GK;
        __syncthreads();
#pragma unroll
        for (int i = 0; i < 4; ++i) {
            int ch  = tid + i * 256;
            int row = ch >> 3;
            int c8  = (ch & 7) << 3;
            const float* sp = &x[(size_t)(mblk + row) * K_DIM + kbase + c8];
            v4f v0 = *reinterpret_cast<const v4f*>(sp);
            v4f v1 = *reinterpret_cast<const v4f*>(sp + 4);
            v4u u;
            u.x = __builtin_bit_cast(unsigned, __builtin_amdgcn_cvt_pkrtz(v0.x, v0.y));
            u.y = __builtin_bit_cast(unsigned, __builtin_amdgcn_cvt_pkrtz(v0.z, v0.w));
            u.z = __builtin_bit_cast(unsigned, __builtin_amdgcn_cvt_pkrtz(v1.x, v1.y));
            u.w = __builtin_bit_cast(unsigned, __builtin_amdgcn_cvt_pkrtz(v1.z, v1.w));
            *reinterpret_cast<v4u*>(&xh[row * F16_STRIDE + c8]) = u;
        }
#endif
        __syncthreads();

        // B fragments: lane L owns weight row n0 + L%16; K-quarter by L/16.
        v16h bfr[2][2];   // [nt][khalf]
#pragma unroll
        for (int nt = 0; nt < 2; ++nt) {
            int r = nblk + wn * 32 + nt * 16 + lmod;
            float s = ws[r * GROUPS + g];
            float b = wb[r * GROUPS + g];
            v2h s2 = {(_Float16)s, (_Float16)s};
            v2h b2 = {(_Float16)b, (_Float16)b};
            const int* base = wp + (size_t)r * WORDS_PER_ROW + g * 16 + lhalf * 4;
            v4i w0 = *reinterpret_cast<const v4i*>(base);       // K {off..off+15}
            v4i w1 = *reinterpret_cast<const v4i*>(base + 8);   // K {32+off..47+off}
            bfr[nt][0] = dequant16(w0, s2, b2);
            bfr[nt][1] = dequant16(w1, s2, b2);
            if (g + 1 < GROUPS) {
                __builtin_prefetch(base + 16, 0, 1);            // next group's words
            }
        }

        // A fragments from LDS + WMMA.
#pragma unroll
        for (int mt = 0; mt < 4; ++mt) {
            int row = wm * 64 + mt * 16 + lmod;
            const _Float16* rb = &xh[row * F16_STRIDE + lhalf * 16];
            v8h a0 = *reinterpret_cast<const v8h*>(rb);
            v8h a1 = *reinterpret_cast<const v8h*>(rb + 8);
            v8h a2 = *reinterpret_cast<const v8h*>(rb + 32);
            v8h a3 = *reinterpret_cast<const v8h*>(rb + 40);
            v16h alo = __builtin_shufflevector(a0, a1,
                0, 1, 2, 3, 4, 5, 6, 7, 8, 9, 10, 11, 12, 13, 14, 15);
            v16h ahi = __builtin_shufflevector(a2, a3,
                0, 1, 2, 3, 4, 5, 6, 7, 8, 9, 10, 11, 12, 13, 14, 15);
#pragma unroll
            for (int nt = 0; nt < 2; ++nt) {
                acc[mt][nt] = __builtin_amdgcn_wmma_f32_16x16x32_f16(
                    false, alo, false, bfr[nt][0], (short)0, acc[mt][nt], false, false);
                acc[mt][nt] = __builtin_amdgcn_wmma_f32_16x16x32_f16(
                    false, ahi, false, bfr[nt][1], (short)0, acc[mt][nt], false, false);
            }
        }
    }

    // Epilogue: + bias, non-temporal f32 stores per C/D layout
    // (VGPR v -> M = v + lhalf*8, N = lmod).
#pragma unroll
    for (int mt = 0; mt < 4; ++mt) {
#pragma unroll
        for (int nt = 0; nt < 2; ++nt) {
            int col = nblk + wn * 32 + nt * 16 + lmod;
            float bv = bias[col];
            int rbase = mblk + wm * 64 + mt * 16 + lhalf * 8;
#pragma unroll
            for (int v = 0; v < 8; ++v) {
                __builtin_nontemporal_store(acc[mt][nt][v] + bv,
                                            &out[(size_t)(rbase + v) * N_DIM + col]);
            }
        }
    }
}

extern "C" void kernel_launch(void* const* d_in, const int* in_sizes, int n_in,
                              void* d_out, int out_size, void* d_ws, size_t ws_size,
                              hipStream_t stream) {
    const float* x    = (const float*)d_in[0];
    const int*   wp   = (const int*)  d_in[1];
    const float* ws   = (const float*)d_in[2];
    const float* wb   = (const float*)d_in[3];
    const float* bias = (const float*)d_in[4];
    float* out = (float*)d_out;

    int Mtot = in_sizes[0] / K_DIM;          // 4096 flattened rows
    dim3 grid(N_DIM / BN, Mtot / BM);        // 32 x 32 workgroups
    gq_linear_wmma<<<grid, 256, 0, stream>>>(x, wp, ws, wb, bias, out);
}
